// IntraMambaBlock_10694468567338
// MI455X (gfx1250) — matrix operand order repaired and checked by reading, not compiled, not run on hardware
//
#include <hip/hip_runtime.h>
#include <hip/hip_bf16.h>

// ---------------------------------------------------------------------------
// Model constants (must match reference)
// ---------------------------------------------------------------------------
#define BN      4
#define TT      1024
#define DM      256
#define DINNER  512
#define DSTATE  16
#define DTRANK  16
#define DCONV   4
#define NHEADS  4
#define DSF     4
#define KS5     5
#define MROWS   (BN * TT)      // 4096
#define TK      (TT / DSF)     // 256 kv timesteps

typedef __attribute__((ext_vector_type(16))) _Float16 v16h;
typedef __attribute__((ext_vector_type(8)))  _Float16 v8h;
typedef __attribute__((ext_vector_type(8)))  float    v8f;
typedef __attribute__((ext_vector_type(4)))  float    v4f;

// ---------------------------------------------------------------------------
// Generic batched WMMA GEMM:  C[M,N] = A[M,K] * B (+bias),  requires K % 32 == 0
//   TRANSB=1 : Bsrc is W[N,K] row-major (dot over K)      -> B(k,n)=W[n,k]
//   TRANSB=0 : Bsrc is V[K,N] row-major                   -> B(k,n)=V[k,n]
// batch z decomposed as (zb, zh) = (z/nh, z%nh) with independent strides.
// Block: 256 threads = 8 waves; tile 128(M) x 64(N) x 32(K);
// each wave computes a 32x32 C block = 4 accumulators, 4 v_wmma per k-step.
// Staging is branchless: out-of-range rows are clamped (their garbage output
// rows/cols are simply never stored by the guarded epilogue).
// ---------------------------------------------------------------------------
#define TM 128
#define TN 64
#define TKK 32
#define APITCH 40   // halves per row (80B, 16B-aligned, bank-conflict padding)
#define BPITCH 40

__device__ inline v16h ld_frag16(const _Float16* p, int o1, int o2) {
    v8h lo = *(const v8h*)(p + o1);
    v8h hi = *(const v8h*)(p + o2);
    v16h r;
#pragma unroll
    for (int j = 0; j < 8; ++j) { r[j] = lo[j]; r[j + 8] = hi[j]; }
    return r;
}

template<int TRANSB, int ADD_BIAS>
__global__ __launch_bounds__(256)
void gemm_wmma(const float* __restrict__ A, const float* __restrict__ Bsrc,
               const float* __restrict__ bias, float* __restrict__ C,
               int M, int N, int K, int lda, int ldb, int ldc,
               int nh, long bA1, long bA2, long bB1, long bB2,
               long bC1, long bC2)
{
    __shared__ __attribute__((aligned(64))) _Float16 Atile[TM * APITCH];
    __shared__ __attribute__((aligned(64))) _Float16 Btile[TN * BPITCH];

    int z  = blockIdx.z;
    int zb = z / nh, zh = z % nh;
    A    += zb * bA1 + zh * bA2;
    Bsrc += zb * bB1 + zh * bB2;
    C    += zb * bC1 + zh * bC2;

    const int m0   = blockIdx.y * TM;
    const int n0   = blockIdx.x * TN;
    const int tid  = threadIdx.x;
    const int lane = tid & 31;
    const int wave = tid >> 5;     // 0..7
    const int wn   = wave & 1;     // 32-col group
    const int wm   = wave >> 1;    // 32-row group
    const int lhalf = lane >> 4;
    const int lrow  = lane & 15;

    // staging assignments (branchless, clamped rows)
    const int ar = tid >> 1;            // 0..127  (A tile row)
    const int ak = (tid & 1) * 16;      // 0,16    (A k chunk of 16)
    const int br = tid >> 2;            // 0..63   (B tile row = n)
    const int bk = (tid & 3) * 8;       // 0,8,16,24

    const int gmA = m0 + ar;
    const long arow = (gmA < M) ? gmA : (M - 1);
    const int gnB = n0 + br;
    const long brow = (gnB < N) ? gnB : (N - 1);

    v8f acc00 = {}, acc01 = {}, acc10 = {}, acc11 = {};

    for (int k0 = 0; k0 < K; k0 += TKK) {
        // ---- stage A tile: 16 contiguous f32 -> f16 per thread ----
        {
            const v4f* ap = (const v4f*)(A + arow * lda + k0 + ak);
            v4f f0 = ap[0], f1 = ap[1], f2 = ap[2], f3 = ap[3];
            v8h h0, h1;
#pragma unroll
            for (int j = 0; j < 4; ++j) {
                h0[j]     = (_Float16)f0[j];
                h0[4 + j] = (_Float16)f1[j];
                h1[j]     = (_Float16)f2[j];
                h1[4 + j] = (_Float16)f3[j];
            }
            *(v8h*)&Atile[ar * APITCH + ak]     = h0;
            *(v8h*)&Atile[ar * APITCH + ak + 8] = h1;
        }
        // ---- stage B tile as [n][k] ----
        {
            v8h hb;
            if (TRANSB) {
                const v4f* bp = (const v4f*)(Bsrc + brow * ldb + k0 + bk);
                v4f g0 = bp[0], g1 = bp[1];
#pragma unroll
                for (int j = 0; j < 4; ++j) {
                    hb[j]     = (_Float16)g0[j];
                    hb[4 + j] = (_Float16)g1[j];
                }
            } else {
#pragma unroll
                for (int j = 0; j < 8; ++j)
                    hb[j] = (_Float16)Bsrc[(long)(k0 + bk + j) * ldb + brow];
            }
            *(v8h*)&Btile[br * BPITCH + bk] = hb;
        }
        __syncthreads();

        // ---- fragments (CDNA5 wmma layouts) ----
        // A 16x32: lanes 0-15 -> K{0..7,16..23}, lanes 16-31 -> K{8..15,24..31}
        v16h a0 = ld_frag16(&Atile[(wm * 32 + lrow) * APITCH],      8 * lhalf, 16 + 8 * lhalf);
        v16h a1 = ld_frag16(&Atile[(wm * 32 + 16 + lrow) * APITCH], 8 * lhalf, 16 + 8 * lhalf);
        // B 32x16: lanes 0-15 = col lrow K{0..7,16..23}, lanes 16-31 K{8..15,24..31}
        v16h b0 = ld_frag16(&Btile[(wn * 32 + lrow) * BPITCH],      8 * lhalf, 16 + 8 * lhalf);
        v16h b1 = ld_frag16(&Btile[(wn * 32 + 16 + lrow) * BPITCH], 8 * lhalf, 16 + 8 * lhalf);

        acc00 = __builtin_amdgcn_wmma_f32_16x16x32_f16(false, a0, false, b0, (short)0, acc00, false, false);
        acc01 = __builtin_amdgcn_wmma_f32_16x16x32_f16(false, a0, false, b1, (short)0, acc01, false, false);
        acc10 = __builtin_amdgcn_wmma_f32_16x16x32_f16(false, a1, false, b0, (short)0, acc10, false, false);
        acc11 = __builtin_amdgcn_wmma_f32_16x16x32_f16(false, a1, false, b1, (short)0, acc11, false, false);
        __syncthreads();
    }

    // ---- epilogue: C f32 layout: VGPR r -> (M = r + 8*lhalf, N = lrow) ----
    int gn0 = n0 + wn * 32 + lrow;
    int gn1 = gn0 + 16;
    float bv0 = 0.f, bv1 = 0.f;
    if (ADD_BIAS) {
        if (gn0 < N) bv0 = bias[gn0];
        if (gn1 < N) bv1 = bias[gn1];
    }
    int mt = m0 + wm * 32 + 8 * lhalf;
#pragma unroll
    for (int r = 0; r < 8; ++r) {
        int gm0 = mt + r;
        int gm1 = mt + 16 + r;
        if (gm0 < M) {
            if (gn0 < N) C[(long)gm0 * ldc + gn0] = acc00[r] + bv0;
            if (gn1 < N) C[(long)gm0 * ldc + gn1] = acc01[r] + bv1;
        }
        if (gm1 < M) {
            if (gn0 < N) C[(long)gm1 * ldc + gn0] = acc10[r] + bv0;
            if (gn1 < N) C[(long)gm1 * ldc + gn1] = acc11[r] + bv1;
        }
    }
}

// ---------------------------------------------------------------------------
// Elementwise / scan kernels
// ---------------------------------------------------------------------------
__device__ inline float dsigmoid(float x) { return 1.f / (1.f + __expf(-x)); }
__device__ inline float dsilu(float x)    { return x * dsigmoid(x); }
__device__ inline float dsoftplus(float x){ return (x > 20.f) ? x : log1pf(__expf(x)); }

// causal depthwise conv (K=4, left pad 3) + SiLU.  xz:[M,1024] -> xc:[M,512]
__global__ __launch_bounds__(256)
void mamba_conv_silu(const float* __restrict__ xz, const float* __restrict__ w,
                     const float* __restrict__ bias, float* __restrict__ xc)
{
    long i = (long)blockIdx.x * blockDim.x + threadIdx.x;
    if (i >= (long)MROWS * DINNER) return;
    int d = (int)(i % DINNER);
    long row = i / DINNER;
    int t = (int)(row % TT);
    long base = row - t;                 // start of this batch
    float s = bias[d];
#pragma unroll
    for (int j = 0; j < DCONV; ++j) {
        int tt = t - (DCONV - 1) + j;
        if (tt >= 0) s += xz[(base + tt) * (2 * DINNER) + d] * w[d * DCONV + j];
    }
    xc[i] = dsilu(s);
}

// Selective scan: one lane per (b,d,n); 16-lane shfl reductions (wave32).
// xdbl:[M,48] = [dt(16) | B(16) | C(16)], delta = softplus(dt@dt_w^T + dt_b)
__global__ __launch_bounds__(256)
void selective_scan(const float* __restrict__ xc, const float* __restrict__ xdbl,
                    const float* __restrict__ dt_w, const float* __restrict__ dt_b,
                    const float* __restrict__ A_log, const float* __restrict__ Dp,
                    float* __restrict__ y)
{
    int tid = blockIdx.x * 256 + threadIdx.x;
    int ch = tid >> 4;          // (b,d)
    int n  = tid & 15;
    if (ch >= BN * DINNER) return;
    int b = ch / DINNER, d = ch % DINNER;

    float Ad  = -__expf(A_log[d * DSTATE + n]);
    float dwn = dt_w[d * DTRANK + n];
    float dtb = dt_b[d];
    float Dd  = Dp[d];
    float h   = 0.f;

    long row = (long)b * TT;
    for (int t = 0; t < TT; ++t, ++row) {
        float u   = xc[row * DINNER + d];
        const float* xr = xdbl + row * 48;
        float dtr = xr[n];
        float Bn  = xr[16 + n];
        float Cn  = xr[32 + n];
        // delta = softplus(sum_r dt[r]*dt_w[d,r] + dt_b[d]) via 16-lane reduce
        float dp = dtr * dwn;
#pragma unroll
        for (int m = 8; m >= 1; m >>= 1) dp += __shfl_xor(dp, m);
        float delta = dsoftplus(dp + dtb);
        float dA = __expf(delta * Ad);
        h = dA * h + delta * u * Bn;
        float yp = h * Cn;
#pragma unroll
        for (int m = 8; m >= 1; m >>= 1) yp += __shfl_xor(yp, m);
        if (n == 0) y[row * DINNER + d] = yp + u * Dd;
    }
}

// y[M,512] *= silu(xz[:,512:1024])
__global__ __launch_bounds__(256)
void gate_silu(float* __restrict__ y, const float* __restrict__ xz)
{
    long i = (long)blockIdx.x * blockDim.x + threadIdx.x;
    if (i >= (long)MROWS * DINNER) return;
    long row = i / DINNER; int c = (int)(i % DINNER);
    y[i] *= dsilu(xz[row * (2 * DINNER) + DINNER + c]);
}

// out[b, T-1-t, c] = in[b, t, c]
__global__ __launch_bounds__(256)
void flip_copy(const float* __restrict__ in, float* __restrict__ out)
{
    long i = (long)blockIdx.x * blockDim.x + threadIdx.x;
    if (i >= (long)MROWS * DM) return;
    long row = i / DM; int c = (int)(i % DM);
    int b = (int)(row / TT), t = (int)(row % TT);
    out[((long)b * TT + (TT - 1 - t)) * DM + c] = in[i];
}

// acc[b, T-1-t, c] += in[b, t, c]
__global__ __launch_bounds__(256)
void add_flip(float* __restrict__ acc, const float* __restrict__ in)
{
    long i = (long)blockIdx.x * blockDim.x + threadIdx.x;
    if (i >= (long)MROWS * DM) return;
    long row = i / DM; int c = (int)(i % DM);
    int b = (int)(row / TT), t = (int)(row % TT);
    acc[((long)b * TT + (TT - 1 - t)) * DM + c] += in[i];
}

// LayerNorm over C=256; one wave32 per row, 8 values/lane. ADD_RES: x+res first.
template<int ADD_RES>
__global__ __launch_bounds__(32)
void layernorm_k(const float* __restrict__ x, const float* __restrict__ res,
                 const float* __restrict__ g, const float* __restrict__ bta,
                 float* __restrict__ out)
{
    long row = blockIdx.x;
    int lane = threadIdx.x;
    float v[8];
    float s = 0.f;
#pragma unroll
    for (int j = 0; j < 8; ++j) {
        long idx = row * DM + lane + 32 * j;
        float t = x[idx];
        if (ADD_RES) t += res[idx];
        v[j] = t; s += t;
    }
#pragma unroll
    for (int m = 16; m >= 1; m >>= 1) s += __shfl_xor(s, m);
    float mean = s * (1.f / DM);
    float q = 0.f;
#pragma unroll
    for (int j = 0; j < 8; ++j) { float d = v[j] - mean; q += d * d; }
#pragma unroll
    for (int m = 16; m >= 1; m >>= 1) q += __shfl_xor(q, m);
    float inv = rsqrtf(q * (1.f / DM) + 1e-5f);
#pragma unroll
    for (int j = 0; j < 8; ++j) {
        int c = lane + 32 * j;
        out[row * DM + c] = (v[j] - mean) * inv * g[c] + bta[c];
    }
}

// in-place softmax over 256 cols with scale; one wave per row
__global__ __launch_bounds__(32)
void softmax256(float* __restrict__ s, float scale)
{
    long row = blockIdx.x;
    int lane = threadIdx.x;
    float v[8];
    float mx = -1e30f;
#pragma unroll
    for (int j = 0; j < 8; ++j) { v[j] = s[row * 256 + lane + 32 * j] * scale; mx = fmaxf(mx, v[j]); }
#pragma unroll
    for (int m = 16; m >= 1; m >>= 1) mx = fmaxf(mx, __shfl_xor(mx, m));
    float sum = 0.f;
#pragma unroll
    for (int j = 0; j < 8; ++j) { v[j] = __expf(v[j] - mx); sum += v[j]; }
#pragma unroll
    for (int m = 16; m >= 1; m >>= 1) sum += __shfl_xor(sum, m);
    float inv = 1.f / sum;
#pragma unroll
    for (int j = 0; j < 8; ++j) s[row * 256 + lane + 32 * j] = v[j] * inv;
}

// depthwise conv K=5, pad 2 (same), over [B,T,DM]
__global__ __launch_bounds__(256)
void dwconv5(const float* __restrict__ x, const float* __restrict__ w,
             const float* __restrict__ bias, float* __restrict__ y)
{
    long i = (long)blockIdx.x * blockDim.x + threadIdx.x;
    if (i >= (long)MROWS * DM) return;
    int c = (int)(i % DM);
    long row = i / DM;
    int t = (int)(row % TT);
    long base = row - t;
    float s = bias[c];
#pragma unroll
    for (int j = 0; j < KS5; ++j) {
        int tt = t + j - KS5 / 2;
        if (tt >= 0 && tt < TT) s += x[(base + tt) * DM + c] * w[c * KS5 + j];
    }
    y[i] = s;
}

__global__ __launch_bounds__(256)
void gelu_k(float* __restrict__ x, long n)
{
    long i = (long)blockIdx.x * blockDim.x + threadIdx.x;
    if (i >= n) return;
    float v = x[i];
    x[i] = 0.5f * v * (1.f + erff(v * 0.70710678118654752f));
}

// red_w[o,c,j] -> wrep[o, j*256+c]  (so KV conv becomes a K=1024 GEMM)
__global__ __launch_bounds__(256)
void repack_redw(const float* __restrict__ rw, float* __restrict__ wrep)
{
    long i = (long)blockIdx.x * blockDim.x + threadIdx.x;
    if (i >= (long)DM * DM * DSF) return;
    int o = (int)(i / (DM * DSF));
    int k = (int)(i % (DM * DSF));
    int j = k / DM, c = k % DM;
    wrep[(long)o * (DM * DSF) + k] = rw[((long)o * DM + c) * DSF + j];
}

// ---------------------------------------------------------------------------
// Host-side helpers
// ---------------------------------------------------------------------------
static void launch_gemm(hipStream_t s, const float* A, const float* B, const float* bias,
                        float* C, int M, int N, int K, int lda, int ldb, int ldc,
                        bool transB, int nz = 1, int nh = 1,
                        long bA1 = 0, long bA2 = 0, long bB1 = 0, long bB2 = 0,
                        long bC1 = 0, long bC2 = 0)
{
    dim3 grid((N + TN - 1) / TN, (M + TM - 1) / TM, nz);
    dim3 blk(256);
    if (transB) {
        if (bias) gemm_wmma<1, 1><<<grid, blk, 0, s>>>(A, B, bias, C, M, N, K, lda, ldb, ldc, nh, bA1, bA2, bB1, bB2, bC1, bC2);
        else      gemm_wmma<1, 0><<<grid, blk, 0, s>>>(A, B, bias, C, M, N, K, lda, ldb, ldc, nh, bA1, bA2, bB1, bB2, bC1, bC2);
    } else {
        if (bias) gemm_wmma<0, 1><<<grid, blk, 0, s>>>(A, B, bias, C, M, N, K, lda, ldb, ldc, nh, bA1, bA2, bB1, bB2, bC1, bC2);
        else      gemm_wmma<0, 0><<<grid, blk, 0, s>>>(A, B, bias, C, M, N, K, lda, ldb, ldc, nh, bA1, bA2, bB1, bB2, bC1, bC2);
    }
}

// p: 0 in_proj[1024,256] 1 conv_w 2 conv_b 3 x_proj[48,512] 4 dt_w 5 dt_b
//    6 A_log 7 D 8 out_proj[256,512]
static void run_mamba(hipStream_t s, const float* xin, const float* const* p, float* out,
                      float* bufA, float* bufB, float* bufC, float* bufD)
{
    // xz = x @ in_proj^T : [4096, 1024]
    launch_gemm(s, xin, p[0], nullptr, bufA, MROWS, 2 * DINNER, DM, DM, DM, 2 * DINNER, true);
    // causal dwconv + silu -> xc [4096, 512]
    mamba_conv_silu<<<(MROWS * DINNER + 255) / 256, 256, 0, s>>>(bufA, p[1], p[2], bufB);
    // x_dbl = xc @ x_proj^T : [4096, 48]
    launch_gemm(s, bufB, p[3], nullptr, bufD, MROWS, DTRANK + 2 * DSTATE, DINNER, DINNER, DINNER, 48, true);
    // selective scan -> y [4096, 512]
    selective_scan<<<(BN * DINNER * DSTATE) / 256, 256, 0, s>>>(bufB, bufD, p[4], p[5], p[6], p[7], bufC);
    // y *= silu(z)
    gate_silu<<<(MROWS * DINNER + 255) / 256, 256, 0, s>>>(bufC, bufA);
    // out = y @ out_proj^T : [4096, 256]
    launch_gemm(s, bufC, p[8], nullptr, out, MROWS, DM, DINNER, DINNER, DINNER, DM, true);
}

extern "C" void kernel_launch(void* const* d_in, const int* in_sizes, int n_in,
                              void* d_out, int out_size, void* d_ws, size_t ws_size,
                              hipStream_t stream)
{
    (void)in_sizes; (void)n_in; (void)out_size; (void)ws_size;
    float* ws = (float*)d_ws;
    size_t o = 0;
    float* bufY = ws + o; o += (size_t)MROWS * DM;        // current activations
    float* bufA = ws + o; o += (size_t)MROWS * 1024;      // xz / attention scores
    float* bufB = ws + o; o += (size_t)MROWS * DINNER;    // xc
    float* bufC = ws + o; o += (size_t)MROWS * DINNER;    // scan y / q
    float* bufD = ws + o; o += (size_t)MROWS * 64;        // x_dbl (ld 48)
    float* bufE = ws + o; o += (size_t)MROWS * DM;        // per-dir mamba out / dw out / AV out
    float* bufF = ws + o; o += (size_t)MROWS * DM;        // bimamba accum / proj out
    float* bufG = ws + o; o += (size_t)MROWS * DM;        // flipped input
    float* kvb  = ws + o; o += (size_t)BN * TK * DM;      // kv (reduced)
    float* kb   = ws + o; o += (size_t)BN * TK * DM;      // K proj
    float* vb   = ws + o; o += (size_t)BN * TK * DM;      // V proj
    float* wrep = ws + o; o += (size_t)DM * DM * DSF;     // repacked red_w

    int idx = 0;
    const float* xin = (const float*)d_in[idx++];
    hipMemcpyAsync(bufY, xin, (size_t)MROWS * DM * sizeof(float),
                   hipMemcpyDeviceToDevice, stream);

    const long elemsMD = (long)MROWS * DM;
    const int  gMD = (int)((elemsMD + 255) / 256);

    for (int layer = 0; layer < 4; ++layer) {
        const float* mf[9]; for (int j = 0; j < 9; ++j) mf[j] = (const float*)d_in[idx++];
        const float* mb[9]; for (int j = 0; j < 9; ++j) mb[j] = (const float*)d_in[idx++];
        const float* ng = (const float*)d_in[idx++];
        const float* nb = (const float*)d_in[idx++];
        const float* ap[8] = {nullptr};
        bool has_attn = (layer % 2 == 0);
        if (has_attn) for (int j = 0; j < 8; ++j) ap[j] = (const float*)d_in[idx++];
        const float* dw[6]; for (int j = 0; j < 6; ++j) dw[j] = (const float*)d_in[idx++];

        // ---- bimamba ----
        run_mamba(stream, bufY, mf, bufF, bufA, bufB, bufC, bufD);
        flip_copy<<<gMD, 256, 0, stream>>>(bufY, bufG);
        run_mamba(stream, bufG, mb, bufE, bufA, bufB, bufC, bufD);
        add_flip<<<gMD, 256, 0, stream>>>(bufF, bufE);
        // y = LN(bimamba + y)
        layernorm_k<1><<<MROWS, 32, 0, stream>>>(bufF, bufY, ng, nb, bufY);

        // ---- attention block (layers 0 and 2) ----
        if (has_attn) {
            // stride-4 reduction conv as GEMM: repack red_w so K index = j*256+c
            repack_redw<<<((long)DM * DM * DSF + 255) / 256, 256, 0, stream>>>(ap[0], wrep);
            // kv[b, tk, :] : per-batch A rows are 4 consecutive x rows (lda=1024)
            launch_gemm(stream, bufY, wrep, ap[1], kvb,
                        TK, DM, DM * DSF, DM * DSF, DM * DSF, DM, true,
                        /*nz=*/BN, /*nh=*/1,
                        (long)TT * DM, 0, 0, 0, (long)TK * DM, 0);
            // q = y @ Wq^T + bq  (Wq = in_w rows 0..255)
            launch_gemm(stream, bufY, ap[2], ap[3], bufC, MROWS, DM, DM, DM, DM, DM, true);
            // k,v from kv
            launch_gemm(stream, kvb, ap[2] + (long)DM * DM,     ap[3] + DM,     kb, BN * TK, DM, DM, DM, DM, DM, true);
            launch_gemm(stream, kvb, ap[2] + (long)2 * DM * DM, ap[3] + 2 * DM, vb, BN * TK, DM, DM, DM, DM, DM, true);
            // scores[b,h] = Q[b,:,h] @ K[b,:,h]^T   (batch 16, dh=64)
            launch_gemm(stream, bufC, kb, nullptr, bufA,
                        TT, TK, DM / NHEADS, DM, DM, TK, true,
                        BN * NHEADS, NHEADS,
                        (long)TT * DM, 64, (long)TK * DM, 64,
                        (long)NHEADS * TT * TK, (long)TT * TK);
            softmax256<<<BN * NHEADS * TT, 32, 0, stream>>>(bufA, 0.125f);
            // O[b,:,h] = S @ V[b,:,h]
            launch_gemm(stream, bufA, vb, nullptr, bufE,
                        TT, DM / NHEADS, TK, TK, DM, DM, false,
                        BN * NHEADS, NHEADS,
                        (long)NHEADS * TT * TK, (long)TT * TK,
                        (long)TK * DM, 64,
                        (long)TT * DM, 64);
            // out proj + residual LN
            launch_gemm(stream, bufE, ap[4], ap[5], bufF, MROWS, DM, DM, DM, DM, DM, true);
            layernorm_k<1><<<MROWS, 32, 0, stream>>>(bufF, bufY, ap[6], ap[7], bufY);
        }

        // ---- dwconv block ----
        dwconv5<<<gMD, 256, 0, stream>>>(bufY, dw[0], dw[1], bufE);
        launch_gemm(stream, bufE, dw[2], dw[3], bufF, MROWS, DM, DM, DM, DM, DM, true);
        gelu_k<<<gMD, 256, 0, stream>>>(bufF, elemsMD);
        layernorm_k<1><<<MROWS, 32, 0, stream>>>(bufF, bufY, dw[4], dw[5], bufY);
    }

    const float* fn_g = (const float*)d_in[idx++];
    const float* fn_b = (const float*)d_in[idx++];
    layernorm_k<0><<<MROWS, 32, 0, stream>>>(bufY, bufY, fn_g, fn_b, (float*)d_out);
}
